// DecoderLayer_8108898255238
// MI455X (gfx1250) — matrix operand order repaired and checked by reading, not compiled, not run on hardware
//
#include <hip/hip_runtime.h>

// ---------------------------------------------------------------------------
// Types / helpers
// ---------------------------------------------------------------------------
typedef __attribute__((ext_vector_type(16))) __bf16        bf16x16;
typedef __attribute__((ext_vector_type(8)))  float         f32x8;
typedef __attribute__((ext_vector_type(4)))  unsigned int  u32x4;
typedef __attribute__((ext_vector_type(4)))  int           i32x4;

#define AS1 __attribute__((address_space(1)))
#define AS3 __attribute__((address_space(3)))

#if __has_builtin(__builtin_amdgcn_global_load_async_to_lds_b128)
#define GEMM_ASYNC 1
#else
#define GEMM_ASYNC 0
#endif

__device__ __forceinline__ void async_copy16(const unsigned short* src,
                                             unsigned short* dst) {
#if GEMM_ASYNC
  __builtin_amdgcn_global_load_async_to_lds_b128((AS1 i32x4*)src,
                                                 (AS3 i32x4*)dst, 0, 0);
#else
  *(u32x4*)dst = *(const u32x4*)src;
#endif
}

__device__ __forceinline__ void wait_async() {
#if __has_builtin(__builtin_amdgcn_s_wait_asynccnt)
  __builtin_amdgcn_s_wait_asynccnt(0);
#elif defined(__AMDGCN__)
  asm volatile("s_wait_asynccnt 0x0" ::: "memory");
#endif
}

__device__ __forceinline__ unsigned short f2bf(float f) {
  union { float f; unsigned u; } c; c.f = f;
  unsigned r = c.u + 0x7FFFu + ((c.u >> 16) & 1u);   // round-to-nearest-even
  return (unsigned short)(r >> 16);
}

constexpr int A_BF16 = 0, A_F32 = 1;
constexpr int B_NT = 0, B_NN = 1;                     // NT: B[N,K] ; NN: B[K,N]
constexpr int EPI_F32 = 0, EPI_F32_RESID = 1, EPI_BF16 = 2, EPI_BF16_ELU = 3;

// ---------------------------------------------------------------------------
// WMMA GEMM:  C[z] = A[z] * B(+bias)(+resid), bf16 inputs, f32 accumulate.
// Block tile 128x256; 8 waves (2x4) of 64x64 each -> 16 wmma / wave / K-chunk.
// ---------------------------------------------------------------------------
template<int ATY, int BLAY, int EPI>
__global__ __launch_bounds__(256) void gemm_wmma_kernel(
    const void* __restrict__ Av, const unsigned short* __restrict__ Bw,
    const float* __restrict__ bias, const float* __restrict__ resid,
    void* __restrict__ Cv,
    int K, int lda, int ldb, int ldc,
    long long strideA, long long strideB, long long strideC, long long strideBias)
{
  constexpr int BM = 128, BN = 256, KC = 32, LK = 40;  // LK pads LDS rows
  __shared__ __attribute__((aligned(16))) unsigned short As[BM * LK];
  __shared__ __attribute__((aligned(16))) unsigned short Bs[BN * LK];

  const int tid  = threadIdx.x;
  const int lane = tid & 31;          // wave32
  const int wave = tid >> 5;
  const int half = lane >> 4;
  const int l16  = lane & 15;
  const int wm   = (wave >> 2) * 64;  // wave row offset inside block tile
  const int wn   = (wave & 3) * 64;   // wave col offset inside block tile
  const long long z = blockIdx.z;
  const int row0 = blockIdx.y * BM;
  const int col0 = blockIdx.x * BN;

  const unsigned short* A16 = (const unsigned short*)Av + ((ATY == A_BF16) ? z * strideA : 0);
  const float*          A32 = (const float*)Av          + ((ATY == A_F32)  ? z * strideA : 0);
  const unsigned short* Bp  = Bw + z * strideB;

  f32x8 acc[4][4];
  {
    f32x8 zv = {0.f,0.f,0.f,0.f,0.f,0.f,0.f,0.f};
    #pragma unroll
    for (int i = 0; i < 4; ++i)
      #pragma unroll
      for (int j = 0; j < 4; ++j) acc[i][j] = zv;
  }

  for (int k0 = 0; k0 < K; k0 += KC) {
    // ---- stage A tile (BM x KC) into LDS, bf16 ----
    {
      const int r = tid >> 1;
      const int c = (tid & 1) << 4;   // 0 or 16
      if (ATY == A_BF16) {
        const unsigned short* src = A16 + (long long)(row0 + r) * lda + k0 + c;
        unsigned short* dst = &As[r * LK + c];
        async_copy16(src,     dst);
        async_copy16(src + 8, dst + 8);
      } else { // f32 -> bf16 convert while staging
        const float* src = A32 + (long long)(row0 + r) * lda + k0 + c;
        unsigned short t[16];
        #pragma unroll
        for (int j = 0; j < 16; ++j) t[j] = f2bf(src[j]);
        u32x4 u0, u1;
        #pragma unroll
        for (int j = 0; j < 4; ++j) u0[j] = (unsigned)t[2*j]   | ((unsigned)t[2*j+1] << 16);
        #pragma unroll
        for (int j = 0; j < 4; ++j) u1[j] = (unsigned)t[8+2*j] | ((unsigned)t[9+2*j] << 16);
        *(u32x4*)&As[r * LK + c]     = u0;
        *(u32x4*)&As[r * LK + c + 8] = u1;
      }
    }
    // ---- stage B tile (BN x KC) into LDS as [n][k] ----
    if (BLAY == B_NT) {
      const int n = tid;              // one full 32-elem row per thread
      const unsigned short* src = Bp + (long long)(col0 + n) * ldb + k0;
      unsigned short* dst = &Bs[n * LK];
      #pragma unroll
      for (int j = 0; j < 4; ++j) async_copy16(src + 8*j, dst + 8*j);
    } else { // B[K,N] -> transpose while storing
      const int kr = tid >> 3;        // 0..31
      const int nb = (tid & 7) << 5;  // 0..224, 32 cols per thread
      const unsigned short* src = Bp + (long long)(k0 + kr) * ldb + col0 + nb;
      u32x4 u[4];
      #pragma unroll
      for (int j = 0; j < 4; ++j) u[j] = *(const u32x4*)(src + 8*j);
      unsigned short t[32];
      #pragma unroll
      for (int j = 0; j < 16; ++j) {
        t[2*j]   = (unsigned short)(u[j >> 2][j & 3] & 0xFFFFu);
        t[2*j+1] = (unsigned short)(u[j >> 2][j & 3] >> 16);
      }
      #pragma unroll
      for (int j = 0; j < 32; ++j) Bs[(nb + j) * LK + kr] = t[j];
    }
    if (GEMM_ASYNC && (ATY == A_BF16 || BLAY == B_NT)) wait_async();
    __syncthreads();

    // ---- fragments (ISA 16-bit A/B wave32 layouts) ----
    union Frag { bf16x16 v; u32x4 u[2]; };
    Frag af[4], bfm[4];
    #pragma unroll
    for (int mi = 0; mi < 4; ++mi) {
      const unsigned short* p = &As[(wm + mi * 16 + l16) * LK + half * 8];
      af[mi].u[0] = *(const u32x4*)p;         // K = klo .. klo+7
      af[mi].u[1] = *(const u32x4*)(p + 16);  // K = klo+16 .. klo+23
    }
    #pragma unroll
    for (int ni = 0; ni < 4; ++ni) {
      const unsigned short* p = &Bs[(wn + ni * 16 + l16) * LK + half * 8];
      bfm[ni].u[0] = *(const u32x4*)p;        // K = klo .. klo+7
      bfm[ni].u[1] = *(const u32x4*)(p + 16); // K = klo+16 .. klo+23
    }
    #pragma unroll
    for (int mi = 0; mi < 4; ++mi)
      #pragma unroll
      for (int ni = 0; ni < 4; ++ni)
        acc[mi][ni] = __builtin_amdgcn_wmma_f32_16x16x32_bf16(
            false, af[mi].v, false, bfm[ni].v, (short)0, acc[mi][ni], false, false);
    __syncthreads();
  }

  // ---- epilogue ----
  #pragma unroll
  for (int mi = 0; mi < 4; ++mi) {
    #pragma unroll
    for (int ni = 0; ni < 4; ++ni) {
      const int gcol = col0 + wn + ni * 16 + l16;
      const float bval = bias ? bias[z * strideBias + gcol] : 0.0f;
      #pragma unroll
      for (int r = 0; r < 8; ++r) {
        const int grow = row0 + wm + mi * 16 + half * 8 + r;
        float vv = acc[mi][ni][r] + bval;
        const long long cidx = z * strideC + (long long)grow * ldc + gcol;
        if (EPI == EPI_F32) {
          ((float*)Cv)[cidx] = vv;
        } else if (EPI == EPI_F32_RESID) {
          ((float*)Cv)[cidx] = vv + resid[(long long)grow * ldc + gcol];
        } else if (EPI == EPI_BF16) {
          ((unsigned short*)Cv)[cidx] = f2bf(vv);
        } else { // EPI_BF16_ELU
          float e = (vv > 0.0f) ? vv : expm1f(vv);
          ((unsigned short*)Cv)[cidx] = f2bf(e);
        }
      }
    }
  }
}

// ---------------------------------------------------------------------------
// LayerNorm over D=512, f32 in -> bf16 out
// ---------------------------------------------------------------------------
__global__ __launch_bounds__(256) void ln_bf16_kernel(
    const float* __restrict__ x, const float* __restrict__ g,
    const float* __restrict__ b, unsigned short* __restrict__ out)
{
  const int D = 512;
  const int row = blockIdx.x;
  const int t = threadIdx.x;
  const float* xr = x + (long long)row * D;
  float v0 = xr[t], v1 = xr[t + 256];
  __shared__ float rs[256], rq[256];
  rs[t] = v0 + v1;
  rq[t] = v0 * v0 + v1 * v1;
  __syncthreads();
  for (int off = 128; off > 0; off >>= 1) {
    if (t < off) { rs[t] += rs[t + off]; rq[t] += rq[t + off]; }
    __syncthreads();
  }
  const float mean = rs[0] * (1.0f / 512.0f);
  const float var  = rq[0] * (1.0f / 512.0f) - mean * mean;
  const float inv  = rsqrtf(var + 1e-5f);
  unsigned short* o = out + (long long)row * D;
  o[t]       = f2bf((v0 - mean) * inv * g[t]       + b[t]);
  o[t + 256] = f2bf((v1 - mean) * inv * g[t + 256] + b[t + 256]);
}

// ---------------------------------------------------------------------------
// Row softmax on scores[h][row][*], in place (f32). Mask folded exactly:
// (-1e20 added before /sqrt(512)) => masked prob == 0.
// ---------------------------------------------------------------------------
__global__ __launch_bounds__(256) void softmax_kernel(float* __restrict__ sc, int causal)
{
  const int L = 1024;
  const int row = blockIdx.x;
  const int h   = blockIdx.y;
  const int t   = threadIdx.x;
  float* s = sc + ((long long)h * L + row) * L;
  const int valid = causal ? (row + 1) : L;
  const float scale = 0.04419417382415922f;  // 1/sqrt(512)
  __shared__ float red[256];

  float m = -3.4e38f;
  for (int c = t; c < valid; c += 256) m = fmaxf(m, s[c] * scale);
  red[t] = m; __syncthreads();
  for (int off = 128; off > 0; off >>= 1) {
    if (t < off) red[t] = fmaxf(red[t], red[t + off]);
    __syncthreads();
  }
  m = red[0]; __syncthreads();

  float sum = 0.0f;
  for (int c = t; c < valid; c += 256) {
    float e = expf(s[c] * scale - m);
    s[c] = e; sum += e;
  }
  red[t] = sum; __syncthreads();
  for (int off = 128; off > 0; off >>= 1) {
    if (t < off) red[t] += red[t + off];
    __syncthreads();
  }
  const float inv = 1.0f / red[0];
  for (int c = t; c < valid; c += 256) s[c] *= inv;
  for (int c = valid + t; c < L; c += 256) s[c] = 0.0f;
}

// ---------------------------------------------------------------------------
// f32 -> bf16 elementwise
// ---------------------------------------------------------------------------
__global__ __launch_bounds__(256) void cvt_kernel(
    const float* __restrict__ in, unsigned short* __restrict__ out, long long n)
{
  long long i = (long long)blockIdx.x * 256 + threadIdx.x;
  if (i < n) out[i] = f2bf(in[i]);
}

// ---------------------------------------------------------------------------
// Host orchestration
// ---------------------------------------------------------------------------
static void attention_block(hipStream_t stream,
    const unsigned short* qsrc, const unsigned short* kvsrc,
    const unsigned short* Wq, const unsigned short* Wk,
    const unsigned short* Wv, const unsigned short* Wc,
    const float* bq, const float* bk, const float* bv, const float* bc,
    unsigned short* qb, unsigned short* kb, unsigned short* vb,
    unsigned short* ob, float* sc, float* xbuf, int causal)
{
  const int L = 1024, D = 512, H = 8, Mtot = 4096, Bb = 4;
  const long long sW = (long long)D * D;       // per-head weight stride
  const long long sQ = (long long)Mtot * D;    // per-head q/k/v stride

  dim3 gProj(D / 256, Mtot / 128, H);
  gemm_wmma_kernel<A_BF16, B_NT, EPI_BF16><<<gProj, 256, 0, stream>>>(
      qsrc, Wq, bq, nullptr, qb, D, D, D, D, 0LL, sW, sQ, (long long)D);
  gemm_wmma_kernel<A_BF16, B_NT, EPI_BF16><<<gProj, 256, 0, stream>>>(
      kvsrc, Wk, bk, nullptr, kb, D, D, D, D, 0LL, sW, sQ, (long long)D);
  gemm_wmma_kernel<A_BF16, B_NT, EPI_BF16><<<gProj, 256, 0, stream>>>(
      kvsrc, Wv, bv, nullptr, vb, D, D, D, D, 0LL, sW, sQ, (long long)D);

  for (int b = 0; b < Bb; ++b) {
    const unsigned short* qh = qb + (long long)b * L * D;
    const unsigned short* kh = kb + (long long)b * L * D;
    const unsigned short* vh = vb + (long long)b * L * D;
    dim3 gS(L / 256, L / 128, H);
    gemm_wmma_kernel<A_BF16, B_NT, EPI_F32><<<gS, 256, 0, stream>>>(
        qh, kh, nullptr, nullptr, sc, D, D, D, L, sQ, sQ, (long long)L * L, 0LL);
    softmax_kernel<<<dim3(L, H), 256, 0, stream>>>(sc, causal);
    dim3 gPV(D / 256, L / 128, H);
    gemm_wmma_kernel<A_F32, B_NN, EPI_BF16><<<gPV, 256, 0, stream>>>(
        sc, vh, nullptr, nullptr, ob + (long long)b * L * (H * D),
        L, L, D, H * D, (long long)L * L, sQ, (long long)D, 0LL);
  }

  dim3 gC(D / 256, Mtot / 128, 1);
  gemm_wmma_kernel<A_BF16, B_NT, EPI_F32_RESID><<<gC, 256, 0, stream>>>(
      ob, Wc, bc, xbuf, xbuf, H * D, H * D, H * D, D, 0LL, 0LL, 0LL, 0LL);
}

extern "C" void kernel_launch(void* const* d_in, const int* in_sizes, int n_in,
                              void* d_out, int out_size, void* d_ws, size_t ws_size,
                              hipStream_t stream) {
  (void)in_sizes; (void)n_in; (void)out_size; (void)ws_size;
  const int Bb = 4, L = 1024, D = 512, H = 8, F = 2048;
  const int Mtot = Bb * L;

  // ---- inputs (setup_inputs dict order, params flattened in insertion order)
  const float* dec   = (const float*)d_in[0];
  const float* enc   = (const float*)d_in[1];
  const float* saWq = (const float*)d_in[2],  *sabq = (const float*)d_in[3];
  const float* saWk = (const float*)d_in[4],  *sabk = (const float*)d_in[5];
  const float* saWv = (const float*)d_in[6],  *sabv = (const float*)d_in[7];
  const float* saWc = (const float*)d_in[8],  *sabc = (const float*)d_in[9];
  const float* caWq = (const float*)d_in[10], *cabq = (const float*)d_in[11];
  const float* caWk = (const float*)d_in[12], *cabk = (const float*)d_in[13];
  const float* caWv = (const float*)d_in[14], *cabv = (const float*)d_in[15];
  const float* caWc = (const float*)d_in[16], *cabc = (const float*)d_in[17];
  const float* fW1  = (const float*)d_in[18], *fb1  = (const float*)d_in[19];
  const float* fW2  = (const float*)d_in[20], *fb2  = (const float*)d_in[21];
  const float* ln1g = (const float*)d_in[22], *ln1b = (const float*)d_in[23];
  const float* ln2g = (const float*)d_in[24], *ln2b = (const float*)d_in[25];
  const float* ln3g = (const float*)d_in[26], *ln3b = (const float*)d_in[27];

  // ---- workspace carve-out
  char* wp = (char*)d_ws;
  auto carve = [&](size_t bytes) -> void* {
    void* p = (void*)wp;
    wp += (bytes + 255) & ~(size_t)255;
    return p;
  };
  float*          xbuf = (float*)carve((size_t)Mtot * D * 4);
  unsigned short* nrm  = (unsigned short*)carve((size_t)Mtot * D * 2);
  unsigned short* encb = (unsigned short*)carve((size_t)Mtot * D * 2);
  unsigned short* qb   = (unsigned short*)carve((size_t)H * Mtot * D * 2);
  unsigned short* kb   = (unsigned short*)carve((size_t)H * Mtot * D * 2);
  unsigned short* vb   = (unsigned short*)carve((size_t)H * Mtot * D * 2);
  unsigned short* ob   = (unsigned short*)carve((size_t)Mtot * H * D * 2);
  unsigned short* hb   = (unsigned short*)carve((size_t)Mtot * F * 2);
  float*          sc   = (float*)carve((size_t)H * L * L * 4);
  unsigned short* wsaWq = (unsigned short*)carve((size_t)H * D * D * 2);
  unsigned short* wsaWk = (unsigned short*)carve((size_t)H * D * D * 2);
  unsigned short* wsaWv = (unsigned short*)carve((size_t)H * D * D * 2);
  unsigned short* wsaWc = (unsigned short*)carve((size_t)D * H * D * 2);
  unsigned short* wcaWq = (unsigned short*)carve((size_t)H * D * D * 2);
  unsigned short* wcaWk = (unsigned short*)carve((size_t)H * D * D * 2);
  unsigned short* wcaWv = (unsigned short*)carve((size_t)H * D * D * 2);
  unsigned short* wcaWc = (unsigned short*)carve((size_t)D * H * D * 2);
  unsigned short* wf1   = (unsigned short*)carve((size_t)F * D * 2);
  unsigned short* wf2   = (unsigned short*)carve((size_t)D * F * 2);

  auto cvt = [&](const float* in, unsigned short* out, long long n) {
    int g = (int)((n + 255) / 256);
    cvt_kernel<<<g, 256, 0, stream>>>(in, out, n);
  };

  // ---- weight / input conversions
  cvt(saWq, wsaWq, (long long)H * D * D);
  cvt(saWk, wsaWk, (long long)H * D * D);
  cvt(saWv, wsaWv, (long long)H * D * D);
  cvt(saWc, wsaWc, (long long)D * H * D);
  cvt(caWq, wcaWq, (long long)H * D * D);
  cvt(caWk, wcaWk, (long long)H * D * D);
  cvt(caWv, wcaWv, (long long)H * D * D);
  cvt(caWc, wcaWc, (long long)D * H * D);
  cvt(fW1, wf1, (long long)F * D);
  cvt(fW2, wf2, (long long)D * F);
  cvt(enc, encb, (long long)Mtot * D);
  (void)hipMemcpyAsync(xbuf, dec, (size_t)Mtot * D * sizeof(float),
                       hipMemcpyDeviceToDevice, stream);

  // ---- 1) masked self-attention + residual
  ln_bf16_kernel<<<Mtot, 256, 0, stream>>>(xbuf, ln1g, ln1b, nrm);
  attention_block(stream, nrm, nrm, wsaWq, wsaWk, wsaWv, wsaWc,
                  sabq, sabk, sabv, sabc, qb, kb, vb, ob, sc, xbuf, /*causal=*/1);

  // ---- 2) cross-attention (K/V from raw encoder embeddings) + residual
  ln_bf16_kernel<<<Mtot, 256, 0, stream>>>(xbuf, ln2g, ln2b, nrm);
  attention_block(stream, nrm, encb, wcaWq, wcaWk, wcaWv, wcaWc,
                  cabq, cabk, cabv, cabc, qb, kb, vb, ob, sc, xbuf, /*causal=*/0);

  // ---- 3) FFN (ELU) + residual -> d_out (f32)
  ln_bf16_kernel<<<Mtot, 256, 0, stream>>>(xbuf, ln3g, ln3b, nrm);
  dim3 g1(F / 256, Mtot / 128, 1);
  gemm_wmma_kernel<A_BF16, B_NT, EPI_BF16_ELU><<<g1, 256, 0, stream>>>(
      nrm, wf1, fb1, nullptr, hb, D, D, D, F, 0LL, 0LL, 0LL, 0LL);
  dim3 g2(D / 256, Mtot / 128, 1);
  gemm_wmma_kernel<A_BF16, B_NT, EPI_F32_RESID><<<g2, 256, 0, stream>>>(
      hb, wf2, fb2, xbuf, (float*)d_out, F, F, F, D, 0LL, 0LL, 0LL, 0LL);
}
